// Attention_59820304498921
// MI455X (gfx1250) — matrix operand order repaired and checked by reading, not compiled, not run on hardware
//
#include <hip/hip_runtime.h>
#include <hip/hip_bf16.h>

typedef __attribute__((ext_vector_type(16))) _Float16 v16h;
typedef __attribute__((ext_vector_type(8)))  _Float16 v8h;
typedef __attribute__((ext_vector_type(8)))  float    v8f;

#define KV_RANK 512
#define ROPE_D  64
#define NOPE_D  128
#define FULL_D  192   // NOPE_D + ROPE_D
#define NH      16
#define SEQ     1024
#define NB      4
#define NTOK    4096  // NB*SEQ
#define SCALE_F 0.07216878364870323f

__device__ inline v8f wmma_f16(v16h a, v16h b, v8f c) {
  return __builtin_amdgcn_wmma_f32_16x16x32_f16(
      /*neg_a=*/false, a, /*neg_b=*/false, b,
      /*c_mod=*/(short)0, c, /*reuse_a=*/false, /*reuse_b=*/false);
}

// ---------------- fragment loaders (wave32 layouts per CDNA5 ISA 7.12.2) ----
// A-matrix 16x32 f16: lanes 0-15 and 16-31 both hold M = lane%16.
// element i<8  -> K = (lane/16)*8 + i
// element i>=8 -> K = 16 + (lane/16)*8 + (i-8)
__device__ inline v16h fragA_f32(const float* src, int stride, int lane, float scale) {
  const int m = lane & 15, hf = lane >> 4;
  const float* p = src + (size_t)m * stride + hf * 8;
  v16h out;
#pragma unroll
  for (int j = 0; j < 8; ++j) {
    out[j]     = (_Float16)(p[j]      * scale);
    out[8 + j] = (_Float16)(p[16 + j] * scale);
  }
  return out;
}

__device__ inline v16h fragA_f16(const _Float16* src, int stride, int lane) {
  const int m = lane & 15, hf = lane >> 4;
  const _Float16* p = src + (size_t)m * stride + hf * 8;
  v8h lo = *(const v8h*)p;
  v8h hi = *(const v8h*)(p + 16);
  v16h out;
#pragma unroll
  for (int j = 0; j < 8; ++j) { out[j] = lo[j]; out[8 + j] = hi[j]; }
  return out;
}

// B-matrix 32x16 f16: lane holds column N = lane%16;
// element i<8 -> K = (lane/16)*8 + i, element i>=8 -> K = 16 + (lane/16)*8 + (i-8).
__device__ inline v16h fragB_f32(const float* src, int stride, int lane) {
  const int n = lane & 15, hf = lane >> 4;
  const float* p = src + (size_t)n * stride + hf * 8;
  v16h out;
#pragma unroll
  for (int i = 0; i < 16; ++i) out[i] = (_Float16)p[i < 8 ? i : i + 8];
  return out;
}

__device__ inline v16h fragB_f16(const _Float16* src, int stride, int lane) {
  const int n = lane & 15, hf = lane >> 4;
  const _Float16* p = src + (size_t)n * stride + hf * 8;
  v8h lo = *(const v8h*)p;
  v8h hi = *(const v8h*)(p + 16);
  v16h out;
#pragma unroll
  for (int j = 0; j < 8; ++j) { out[j] = lo[j]; out[8 + j] = hi[j]; }
  return out;
}

// ---------------------------------------------------------------------------
// Kernel 1: latent projection.
//   kf16[n][h][0:128]   = kv_c[n] . w_key[h]^T   (f16)
//   kf16[n][h][128:192] = k_rope[n]              (f16, broadcast over h)
//   vT[h][d][n]         = kv_c[n] . w_vo[h]^T    (f16, key-major for PV B-frags)
// grid: (NTOK/16, NH), block: 256 (8 waves, each owns 16 output dims)
// ---------------------------------------------------------------------------
__global__ void __launch_bounds__(256)
mla_proj_kernel(const float* __restrict__ kin,   // (NTOK, 576)
                const float* __restrict__ wkey,  // (NH, 128, 512)
                const float* __restrict__ wvo,   // (NH, 128, 512)
                _Float16* __restrict__ kf,       // (NTOK, NH, 192)
                _Float16* __restrict__ vT) {     // (NH, 128, NTOK)
  const int n0   = blockIdx.x * 16;
  const int h    = blockIdx.y;
  const int wave = threadIdx.x >> 5;
  const int lane = threadIdx.x & 31;
  const int d0   = wave * 16;

  const float* abase = kin  + (size_t)n0 * 576;
  const float* wk    = wkey + ((size_t)h * 128 + d0) * 512;
  const float* wv    = wvo  + ((size_t)h * 128 + d0) * 512;

  v8f ck = {}, cv = {};
#pragma unroll 4
  for (int k0 = 0; k0 < KV_RANK; k0 += 32) {
    v16h a  = fragA_f32(abase + k0, 576, lane, 1.0f);
    v16h bk = fragB_f32(wk + k0, 512, lane);
    v16h bv = fragB_f32(wv + k0, 512, lane);
    ck = wmma_f16(a, bk, ck);
    cv = wmma_f16(a, bv, cv);
  }

  const int hf = lane >> 4, nc = lane & 15;
#pragma unroll
  for (int r = 0; r < 8; ++r) {
    const int n = n0 + r + 8 * hf;
    kf[((size_t)n * NH + h) * FULL_D + d0 + nc]   = (_Float16)ck[r];
    vT[((size_t)h * 128 + d0 + nc) * NTOK + n]    = (_Float16)cv[r];
  }

  // broadcast rope dims into kf16 cols 128..191 for this head
  for (int i = threadIdx.x; i < 16 * ROPE_D; i += 256) {
    const int row = i >> 6, col = i & 63;
    kf[((size_t)(n0 + row) * NH + h) * FULL_D + NOPE_D + col] =
        (_Float16)kin[(size_t)(n0 + row) * 576 + KV_RANK + col];
  }
}

// ---------------------------------------------------------------------------
// Kernel 2: causal flash attention, one 16-row q-tile per wave.
// grid: (SEQ/128, NH, NB), block 256 (8 waves)
// ---------------------------------------------------------------------------
__global__ void __launch_bounds__(256)
mla_attn_kernel(const float* __restrict__ qin,       // (NTOK, NH, 192)
                const _Float16* __restrict__ kf,     // (NTOK, NH, 192)
                const _Float16* __restrict__ vT,     // (NH, 128, NTOK)
                float* __restrict__ out) {           // (NTOK, NH, 128)
  __shared__ _Float16 plds[8][16 * 32];

  const int wave = threadIdx.x >> 5;
  const int lane = threadIdx.x & 31;
  const int hf   = lane >> 4;
  const int nc   = lane & 15;
  const int h    = blockIdx.y;
  const int b    = blockIdx.z;
  const int q0   = blockIdx.x * 128 + wave * 16;  // q-tile base within sequence
  const size_t bS = (size_t)b * SEQ;
  const int kstride = NH * FULL_D;                // row stride in kf16 (f16)

  // Q fragments, SCALE folded into the f16 conversion
  v16h qfrag[6];
  const float* qbase = qin + ((bS + q0) * NH + h) * FULL_D;
#pragma unroll
  for (int dc = 0; dc < 6; ++dc)
    qfrag[dc] = fragA_f32(qbase + dc * 32, kstride, lane, SCALE_F);

  float m_run[8], l_run[8], corr[8];
  v8f o_acc[8];
  const v8f vzero = {};
#pragma unroll
  for (int r = 0; r < 8; ++r) { m_run[r] = -1e30f; l_run[r] = 0.0f; }
#pragma unroll
  for (int t = 0; t < 8; ++t) o_acc[t] = vzero;

  for (int k0 = 0; k0 < q0 + 16; k0 += 32) {
    const _Float16* kb = kf + ((bS + k0) * NH + h) * FULL_D;
    __builtin_prefetch(kb + (size_t)32 * kstride, 0, 1);  // next key tile

    // ---- scores: S = (scaled Q) . K^T, two 16-key subtiles -----------------
    v8f s0 = vzero, s1 = vzero;
#pragma unroll
    for (int dc = 0; dc < 6; ++dc) {
      v16h b0 = fragB_f16(kb + dc * 32, kstride, lane);
      v16h b1 = fragB_f16(kb + (size_t)16 * kstride + dc * 32, kstride, lane);
      s0 = wmma_f16(qfrag[dc], b0, s0);
      s1 = wmma_f16(qfrag[dc], b1, s1);
    }

    // ---- causal mask + online softmax (row stats live per lane) ------------
#pragma unroll
    for (int r = 0; r < 8; ++r) {
      const int qrow = q0 + r + 8 * hf;
      if (k0 + nc      > qrow) s0[r] = -1e30f;
      if (k0 + 16 + nc > qrow) s1[r] = -1e30f;

      float mx = fmaxf(s0[r], s1[r]);
#pragma unroll
      for (int off = 1; off < 16; off <<= 1)
        mx = fmaxf(mx, __shfl_xor(mx, off, 32));

      const float m_new = fmaxf(m_run[r], mx);
      corr[r] = __expf(m_run[r] - m_new);
      const float p0 = __expf(s0[r] - m_new);
      const float p1 = __expf(s1[r] - m_new);
      float rs = p0 + p1;
#pragma unroll
      for (int off = 1; off < 16; off <<= 1)
        rs += __shfl_xor(rs, off, 32);
      l_run[r] = l_run[r] * corr[r] + rs;
      m_run[r] = m_new;

      const int row = r + 8 * hf;
      plds[wave][row * 32 + nc]      = (_Float16)p0;
      plds[wave][row * 32 + 16 + nc] = (_Float16)p1;
    }

    // ---- O = diag(corr)*O + P . V  (K = 32 keys per step) ------------------
    v16h pf = fragA_f16(&plds[wave][0], 32, lane);
    const _Float16* vb = vT + bS + k0;
#pragma unroll
    for (int t = 0; t < 8; ++t) {
#pragma unroll
      for (int r = 0; r < 8; ++r) o_acc[t][r] *= corr[r];
      v16h bv = fragB_f16(vb + ((size_t)h * 128 + t * 16) * NTOK, NTOK, lane);
      o_acc[t] = wmma_f16(pf, bv, o_acc[t]);
    }
  }

  // ---- normalize and store f32 output --------------------------------------
#pragma unroll
  for (int t = 0; t < 8; ++t) {
#pragma unroll
    for (int r = 0; r < 8; ++r) {
      const int qrow = q0 + r + 8 * hf;
      out[((bS + qrow) * NH + h) * NOPE_D + t * 16 + nc] = o_acc[t][r] / l_run[r];
    }
  }
}

// ---------------------------------------------------------------------------
extern "C" void kernel_launch(void* const* d_in, const int* in_sizes, int n_in,
                              void* d_out, int out_size, void* d_ws, size_t ws_size,
                              hipStream_t stream) {
  const float* q     = (const float*)d_in[0];
  const float* k     = (const float*)d_in[1];
  // d_in[2] (v) is unused by the reference: v_raw is derived from kv_c.
  const float* w_key = (const float*)d_in[3];
  const float* w_vo  = (const float*)d_in[4];

  _Float16* kf16 = (_Float16*)d_ws;                                    // 25.2 MB
  _Float16* vT   = (_Float16*)((char*)d_ws +
                    (size_t)NTOK * NH * FULL_D * sizeof(_Float16));    // +16.8 MB

  dim3 g1(NTOK / 16, NH);
  mla_proj_kernel<<<g1, dim3(256), 0, stream>>>(k, w_key, w_vo, kf16, vT);

  dim3 g2(SEQ / 128, NH, NB);
  mla_attn_kernel<<<g2, dim3(256), 0, stream>>>(q, kf16, vT, (float*)d_out);
}
